// QwenAttention_17858474017046
// MI455X (gfx1250) — matrix operand order repaired and checked by reading, not compiled, not run on hardware
//
#include <hip/hip_runtime.h>
#include <hip/hip_bf16.h>
#include <cstddef>

// ---------------- problem constants ----------------
constexpr int S  = 4096;
constexpr int D  = 896;
constexpr int NH = 14;
constexpr int NKV = 2;
constexpr int HD = 64;
constexpr float ROPE_BASE = 1000000.0f;

// ---------------- vector types ----------------
typedef __attribute__((ext_vector_type(16))) _Float16 v16h;
typedef __attribute__((ext_vector_type(8)))  _Float16 v8h;
typedef __attribute__((ext_vector_type(8)))  float    v8f;

static __device__ __forceinline__ v16h join8(v8h lo, v8h hi) {
  return __builtin_shufflevector(lo, hi, 0,1,2,3,4,5,6,7,8,9,10,11,12,13,14,15);
}

static __device__ __forceinline__ v8f wmma_f16(v16h a, v16h b, v8f c) {
  return __builtin_amdgcn_wmma_f32_16x16x32_f16(false, a, false, b, (short)0, c, false, false);
}

// ---- DPP16 row_xmask butterfly (lane ^ MASK within 16-lane rows): pure VALU, no DS ----
template <int MASK>
static __device__ __forceinline__ float bfly_max(float x) {
  float y = __int_as_float(__builtin_amdgcn_update_dpp(
      0, __float_as_int(x), 0x160 | MASK, 0xf, 0xf, true));
  return fmaxf(x, y);
}
template <int MASK>
static __device__ __forceinline__ float bfly_add(float x) {
  float y = __int_as_float(__builtin_amdgcn_update_dpp(
      0, __float_as_int(x), 0x160 | MASK, 0xf, 0xf, true));
  return x + y;
}

// ---------------- elementwise converts ----------------
__global__ void cvt16_kernel(const float* __restrict__ X, _Float16* __restrict__ Y, int n) {
  int i = blockIdx.x * blockDim.x + threadIdx.x;
  if (i < n) Y[i] = (_Float16)X[i];
}

// W [K][N] fp32 -> Wt [N][K] f16 (K-major columns => contiguous B fragments)
__global__ void transpose_w_kernel(const float* __restrict__ W, _Float16* __restrict__ Wt,
                                   int K, int N) {
  int i = blockIdx.x * blockDim.x + threadIdx.x;
  if (i >= K * N) return;
  int n = i % N;
  int k = i / N;
  Wt[(size_t)n * K + k] = (_Float16)W[(size_t)k * N + n];
}

// ---------------- WMMA GEMM: C[M][N] = A[M][K] * Bt[N][K]^T + bias ----------------
// One wave computes a 32x64 tile: two A fragments share the same four B fragments
// (8 WMMAs per 192B/lane of loads instead of 4 per 160B).
template <typename OutT>
__global__ void gemm_wmma_kernel(const _Float16* __restrict__ A,
                                 const _Float16* __restrict__ Bt,
                                 const float* __restrict__ bias,
                                 OutT* __restrict__ C,
                                 int M, int N, int K) {
  int wave = (int)((blockIdx.x * blockDim.x + threadIdx.x) >> 5);
  int lane = threadIdx.x & 31;
  int l  = lane & 15;
  int hw = lane >> 4;
  int Nt = N >> 6;
  int Mt = M >> 5;
  if (wave >= Mt * Nt) return;
  int tn = wave % Nt, tm = wave / Nt;
  int mBase = tm * 32, nBase = tn * 64;

  v8f acc[2][4];
  #pragma unroll
  for (int g = 0; g < 2; ++g)
    #pragma unroll
    for (int t = 0; t < 4; ++t) {
      v8f z = {0.f,0.f,0.f,0.f,0.f,0.f,0.f,0.f};
      acc[g][t] = z;
    }

  // A fragment: lane holds row mBase+l (+16), K piece (hw*16..hw*16+15)
  const _Float16* aRow0 = A + (size_t)(mBase + l) * K + hw * 16;
  const _Float16* aRow1 = aRow0 + (size_t)16 * K;
  for (int k0 = 0; k0 < K; k0 += 32) {
    v16h af0 = join8(*(const v8h*)(aRow0 + k0), *(const v8h*)(aRow0 + k0 + 8));
    v16h af1 = join8(*(const v8h*)(aRow1 + k0), *(const v8h*)(aRow1 + k0 + 8));
    #pragma unroll
    for (int t = 0; t < 4; ++t) {
      const _Float16* bp = Bt + (size_t)(nBase + t * 16 + l) * K + k0 + hw * 16;
      v16h bf = join8(*(const v8h*)bp, *(const v8h*)(bp + 8));
      acc[0][t] = wmma_f16(af0, bf, acc[0][t]);
      acc[1][t] = wmma_f16(af1, bf, acc[1][t]);
    }
  }

  #pragma unroll
  for (int t = 0; t < 4; ++t) {
    int n = nBase + t * 16 + l;
    float bvv = bias ? bias[n] : 0.0f;
    #pragma unroll
    for (int g = 0; g < 2; ++g) {
      #pragma unroll
      for (int r = 0; r < 8; ++r) {
        int m = mBase + g * 16 + r + 8 * hw;
        C[(size_t)m * N + n] = (OutT)(acc[g][t][r] + bvv);
      }
    }
  }
}

// ---------------- RoPE + repack ----------------
// X [S][nh*HD] f16 -> Y [nh][S][HD] f16 with RoPE applied
__global__ void rope_pack_kernel(const _Float16* __restrict__ X, _Float16* __restrict__ Y,
                                 int nh, int ldx) {
  int idx = blockIdx.x * blockDim.x + threadIdx.x;
  int total = nh * S * 32;
  if (idx >= total) return;
  int j = idx & 31;
  int s = (idx >> 5) % S;
  int h = idx / (32 * S);
  float invf = __expf(-((float)(2 * j) / (float)HD) * __logf(ROPE_BASE));
  float ang = (float)s * invf;
  float sn, cs;
  __sincosf(ang, &sn, &cs);
  float x0 = (float)X[(size_t)s * ldx + h * HD + j];
  float x1 = (float)X[(size_t)s * ldx + h * HD + j + 32];
  size_t ob = ((size_t)h * S + s) * HD;
  Y[ob + j]      = (_Float16)(x0 * cs - x1 * sn);
  Y[ob + j + 32] = (_Float16)(x1 * cs + x0 * sn);
}

// X [S][NKV*HD] f16 -> Y [NKV][HD][S] f16 (transposed so PV B-fragments are contiguous)
__global__ void pack_v_kernel(const _Float16* __restrict__ X, _Float16* __restrict__ Y) {
  int idx = blockIdx.x * blockDim.x + threadIdx.x;
  int total = NKV * HD * S;
  if (idx >= total) return;
  int s = idx % S;
  int d = (idx / S) % HD;
  int kv = idx / (S * HD);
  Y[idx] = X[(size_t)s * (NKV * HD) + kv * HD + d];
}

// ---------------- Flash attention (causal, GQA), one wave per 16-query tile ----------------
__global__ void attn_wmma_kernel(const _Float16* __restrict__ Q,  // [NH][S][HD]
                                 const _Float16* __restrict__ Km, // [NKV][S][HD]
                                 const _Float16* __restrict__ Vt, // [NKV][HD][S]
                                 _Float16* __restrict__ O)        // [S][NH*HD]
{
  __shared__ __align__(16) _Float16 pbuf[8][16 * 32];  // per-wave P tile (C->A layout bridge)
  int waveInBlk = threadIdx.x >> 5;
  int wave = blockIdx.x * (blockDim.x >> 5) + waveInBlk;
  int lane = threadIdx.x & 31;
  int l  = lane & 15;
  int hw = lane >> 4;
  int total = NH * (S / 16);
  if (wave >= total) return;
  int h  = wave / (S / 16);
  int qt = wave % (S / 16);
  int qBase = qt * 16;
  int kv = h / (NH / NKV);

  const _Float16* Qh = Q + ((size_t)h * S + qBase) * HD;
  const _Float16* Kh = Km + (size_t)kv * S * HD;
  const _Float16* Vh = Vt + (size_t)kv * HD * S;
  _Float16* lds = &pbuf[waveInBlk][0];

  // Q A-fragments for d 0..31 and 32..63
  const _Float16* qrow = Qh + (size_t)l * HD;
  v16h qa0 = join8(*(const v8h*)(qrow + hw * 16),      *(const v8h*)(qrow + 8 + hw * 16));
  v16h qa1 = join8(*(const v8h*)(qrow + 32 + hw * 16), *(const v8h*)(qrow + 40 + hw * 16));

  float mrow[8], lrow[8];
  v8f acc[4];
  #pragma unroll
  for (int r = 0; r < 8; ++r) { mrow[r] = -1e30f; lrow[r] = 0.f; }
  #pragma unroll
  for (int t = 0; t < 4; ++t) {
    v8f z = {0.f,0.f,0.f,0.f,0.f,0.f,0.f,0.f};
    acc[t] = z;
  }

  const float scale = 0.125f;  // 1/sqrt(64)
  int qEnd = qBase + 15;
  for (int kb = 0; kb <= qEnd; kb += 32) {
    // --- scores: two 16-key subtiles, K-dim = 64 split into two WMMAs each ---
    v8f c0 = {0.f,0.f,0.f,0.f,0.f,0.f,0.f,0.f};
    v8f c1 = {0.f,0.f,0.f,0.f,0.f,0.f,0.f,0.f};
    {
      const _Float16* kr0 = Kh + (size_t)(kb + l) * HD + hw * 16;
      v16h b00 = join8(*(const v8h*)(kr0),      *(const v8h*)(kr0 + 8));
      v16h b01 = join8(*(const v8h*)(kr0 + 32), *(const v8h*)(kr0 + 40));
      c0 = wmma_f16(qa0, b00, c0);
      c0 = wmma_f16(qa1, b01, c0);
      const _Float16* kr1 = kr0 + (size_t)16 * HD;
      v16h b10 = join8(*(const v8h*)(kr1),      *(const v8h*)(kr1 + 8));
      v16h b11 = join8(*(const v8h*)(kr1 + 32), *(const v8h*)(kr1 + 40));
      c1 = wmma_f16(qa0, b10, c1);
      c1 = wmma_f16(qa1, b11, c1);
    }
    // --- causal mask + online softmax (row = r + 8*hw, col = l within subtile) ---
    #pragma unroll
    for (int r = 0; r < 8; ++r) {
      int qi = qBase + r + 8 * hw;
      float s0 = (kb + l      <= qi) ? c0[r] * scale : -1e30f;
      float s1 = (kb + 16 + l <= qi) ? c1[r] * scale : -1e30f;
      // DPP row_xmask butterfly over the 16-lane half (masks 1,2,4,8)
      float t = fmaxf(s0, s1);
      t = bfly_max<1>(t);
      t = bfly_max<2>(t);
      t = bfly_max<4>(t);
      t = bfly_max<8>(t);
      float mnew = fmaxf(mrow[r], t);
      float sc = __expf(mrow[r] - mnew);
      float p0 = __expf(s0 - mnew);
      float p1 = __expf(s1 - mnew);
      float rs = p0 + p1;
      rs = bfly_add<1>(rs);
      rs = bfly_add<2>(rs);
      rs = bfly_add<4>(rs);
      rs = bfly_add<8>(rs);
      lrow[r] = lrow[r] * sc + rs;
      mrow[r] = mnew;
      #pragma unroll
      for (int t4 = 0; t4 < 4; ++t4) acc[t4][r] *= sc;
      int row = r + 8 * hw;
      lds[row * 32 + l]      = (_Float16)p0;
      lds[row * 32 + 16 + l] = (_Float16)p1;
    }
    // intra-wave DS ordering: stores complete before A-layout reload
    asm volatile("s_wait_dscnt 0x0" ::: "memory");
    const _Float16* prow = lds + l * 32 + hw * 16;
    v16h pa = join8(*(const v8h*)prow, *(const v8h*)(prow + 8));
    // --- P (16x32) @ V (32x64): 4 output d-subtiles ---
    #pragma unroll
    for (int t4 = 0; t4 < 4; ++t4) {
      const _Float16* vp = Vh + (size_t)(t4 * 16 + l) * S + kb + hw * 16;
      v16h vb = join8(*(const v8h*)vp, *(const v8h*)(vp + 8));
      acc[t4] = wmma_f16(pa, vb, acc[t4]);
    }
    asm volatile("s_wait_dscnt 0x0" ::: "memory");
  }

  // epilogue: normalize, write f16 [S][NH*HD]
  #pragma unroll
  for (int r = 0; r < 8; ++r) {
    float inv = 1.0f / lrow[r];
    int m = qBase + r + 8 * hw;
    #pragma unroll
    for (int t4 = 0; t4 < 4; ++t4) {
      O[(size_t)m * (NH * HD) + h * HD + t4 * 16 + l] = (_Float16)(acc[t4][r] * inv);
    }
  }
}

// ---------------- host launch ----------------
extern "C" void kernel_launch(void* const* d_in, const int* in_sizes, int n_in,
                              void* d_out, int out_size, void* d_ws, size_t ws_size,
                              hipStream_t stream) {
  (void)in_sizes; (void)n_in; (void)out_size; (void)ws_size;
  const float* hid = (const float*)d_in[0];
  // d_in[1] = attention_mask: pure causal, computed analytically -> never read (64MB saved)
  const float* Wq = (const float*)d_in[2];
  const float* bq = (const float*)d_in[3];
  const float* Wk = (const float*)d_in[4];
  const float* bk = (const float*)d_in[5];
  const float* Wv = (const float*)d_in[6];
  const float* bv = (const float*)d_in[7];
  const float* Wo = (const float*)d_in[8];
  float* out = (float*)d_out;

  char* ws = (char*)d_ws;
  size_t off = 0;
  auto alloc = [&](size_t bytes) -> char* {
    char* p = ws + off;
    off = (off + bytes + 255) & ~(size_t)255;
    return p;
  };

  _Float16* h16   = (_Float16*)alloc((size_t)S * D * 2);
  _Float16* wq_t  = (_Float16*)alloc((size_t)D * D * 2);          // [896][896]
  _Float16* wk_t  = (_Float16*)alloc((size_t)(NKV * HD) * D * 2); // [128][896]
  _Float16* wv_t  = (_Float16*)alloc((size_t)(NKV * HD) * D * 2);
  _Float16* wo_t  = (_Float16*)alloc((size_t)D * D * 2);
  _Float16* q16   = (_Float16*)alloc((size_t)S * (NH * HD) * 2);
  _Float16* k16   = (_Float16*)alloc((size_t)S * (NKV * HD) * 2);
  _Float16* v16   = (_Float16*)alloc((size_t)S * (NKV * HD) * 2);
  _Float16* Qr    = (_Float16*)alloc((size_t)NH * S * HD * 2);
  _Float16* Kr    = (_Float16*)alloc((size_t)NKV * S * HD * 2);
  _Float16* Vtr   = (_Float16*)alloc((size_t)NKV * HD * S * 2);
  _Float16* attn  = (_Float16*)alloc((size_t)S * (NH * HD) * 2);

  const int TB = 256;
  auto blocks = [](long n, int tb) { return (int)((n + tb - 1) / tb); };

  // 1) converts / transposes
  cvt16_kernel<<<blocks((long)S * D, TB), TB, 0, stream>>>(hid, h16, S * D);
  transpose_w_kernel<<<blocks((long)D * D, TB), TB, 0, stream>>>(Wq, wq_t, D, NH * HD);
  transpose_w_kernel<<<blocks((long)D * NKV * HD, TB), TB, 0, stream>>>(Wk, wk_t, D, NKV * HD);
  transpose_w_kernel<<<blocks((long)D * NKV * HD, TB), TB, 0, stream>>>(Wv, wv_t, D, NKV * HD);
  transpose_w_kernel<<<blocks((long)D * D, TB), TB, 0, stream>>>(Wo, wo_t, D, D);

  // 2) QKV projections (WMMA, f16 out); 32x64 tile per wave
  {
    int waves = (S / 32) * ((NH * HD) / 64);
    gemm_wmma_kernel<_Float16><<<blocks((long)waves * 32, TB), TB, 0, stream>>>(
        h16, wq_t, bq, q16, S, NH * HD, D);
  }
  {
    int waves = (S / 32) * ((NKV * HD) / 64);
    gemm_wmma_kernel<_Float16><<<blocks((long)waves * 32, TB), TB, 0, stream>>>(
        h16, wk_t, bk, k16, S, NKV * HD, D);
    gemm_wmma_kernel<_Float16><<<blocks((long)waves * 32, TB), TB, 0, stream>>>(
        h16, wv_t, bv, v16, S, NKV * HD, D);
  }

  // 3) RoPE + layout repack
  rope_pack_kernel<<<blocks((long)NH * S * 32, TB), TB, 0, stream>>>(q16, Qr, NH, NH * HD);
  rope_pack_kernel<<<blocks((long)NKV * S * 32, TB), TB, 0, stream>>>(k16, Kr, NKV, NKV * HD);
  pack_v_kernel<<<blocks((long)NKV * HD * S, TB), TB, 0, stream>>>(v16, Vtr);

  // 4) causal GQA flash attention
  {
    int waves = NH * (S / 16);
    attn_wmma_kernel<<<blocks((long)waves * 32, TB), TB, 0, stream>>>(Qr, Kr, Vtr, attn);
  }

  // 5) output projection (WMMA, f32 out -> d_out)
  {
    int waves = (S / 32) * (D / 64);
    gemm_wmma_kernel<float><<<blocks((long)waves * 32, TB), TB, 0, stream>>>(
        attn, wo_t, nullptr, out, S, D, D);
  }
}
